// mixture_of_experts_32263794327932
// MI455X (gfx1250) — matrix operand order repaired and checked by reading, not compiled, not run on hardware
//
#include <hip/hip_runtime.h>
#include <hip/hip_bf16.h>
#include <math.h>

// Problem constants (from reference): B=2,S=512,H=768,I=3072,E=8,K=1
#define T_TOK 1024
#define HDIM  768
#define IDIM  3072
#define NEXP  8

typedef _Float16 v16h __attribute__((ext_vector_type(16)));
typedef _Float16 h8   __attribute__((ext_vector_type(8)));
typedef float    v8f  __attribute__((ext_vector_type(8)));

// Convert 16 consecutive fp32 (one B-fragment lane slice) to a v16h.
__device__ __forceinline__ v16h load_b_frag_f32(const float* p) {
    const float4* pb = (const float4*)p;
    float4 f0 = pb[0], f1 = pb[1], f2 = pb[2], f3 = pb[3];
    v16h b;
    b[0]=(_Float16)f0.x;  b[1]=(_Float16)f0.y;  b[2]=(_Float16)f0.z;  b[3]=(_Float16)f0.w;
    b[4]=(_Float16)f1.x;  b[5]=(_Float16)f1.y;  b[6]=(_Float16)f1.z;  b[7]=(_Float16)f1.w;
    b[8]=(_Float16)f2.x;  b[9]=(_Float16)f2.y;  b[10]=(_Float16)f2.z; b[11]=(_Float16)f2.w;
    b[12]=(_Float16)f3.x; b[13]=(_Float16)f3.y; b[14]=(_Float16)f3.z; b[15]=(_Float16)f3.w;
    return b;
}

// Load the 16x32 f16 A-fragment slice for this lane from an LDS row.
// Layout (contiguous 16-bit A 16x32): lane = M + 16*(K/16), element = K%16.
__device__ __forceinline__ v16h load_a_frag_lds(const _Float16* row, int k0, int ghi) {
    h8 lo = *(const h8*)(row + k0 + 16 * ghi);
    h8 hi = *(const h8*)(row + k0 + 16 * ghi + 8);
    v16h a;
#pragma unroll
    for (int q = 0; q < 8; ++q) { a[q] = lo[q]; a[q + 8] = hi[q]; }
    return a;
}

// ---------------------------------------------------------------------------
// Kernel 0: zero the routing counters (ws is poisoned 0xAA each timing pass).
// ---------------------------------------------------------------------------
__global__ void moe_init_kernel(int* count, int* cursor) {
    int t = threadIdx.x;
    if (t < NEXP) { count[t] = 0; cursor[t] = 0; }
}

// ---------------------------------------------------------------------------
// Kernel 1: gate logits + top-1 routing. One wave (32 lanes) per token.
// ---------------------------------------------------------------------------
__global__ __launch_bounds__(32)
void moe_gate_kernel(const float* __restrict__ x, const float* __restrict__ gw,
                     const float* __restrict__ gb, int* __restrict__ sel,
                     int* __restrict__ count, float* __restrict__ logits) {
    int t = blockIdx.x;
    int lane = threadIdx.x;
    float acc[NEXP];
#pragma unroll
    for (int e = 0; e < NEXP; ++e) acc[e] = 0.0f;
    const float* xr = x + t * HDIM;
    for (int k = lane; k < HDIM; k += 32) {
        float xv = xr[k];
#pragma unroll
        for (int e = 0; e < NEXP; ++e) acc[e] += xv * gw[e * HDIM + k];
    }
#pragma unroll
    for (int e = 0; e < NEXP; ++e) {
#pragma unroll
        for (int off = 16; off > 0; off >>= 1)
            acc[e] += __shfl_xor(acc[e], off, 32);
    }
    if (lane == 0) {
        float best = -INFINITY; int bi = 0;
#pragma unroll
        for (int e = 0; e < NEXP; ++e) {
            float lg = acc[e] + gb[e];
            logits[t * NEXP + e] = lg;
            if (lg > best) { best = lg; bi = e; }   // strict '>' == top_k tie-break
        }
        sel[t] = bi;
        atomicAdd(&count[bi], 1);
    }
}

// ---------------------------------------------------------------------------
// Kernel 2: load-balance loss (deterministic reduction: one wave per expert)
// plus expert offsets and 16-token group table.
// ---------------------------------------------------------------------------
__global__ __launch_bounds__(256)
void moe_plan_kernel(const float* __restrict__ logits, const int* __restrict__ count,
                     int* __restrict__ offs, int* __restrict__ gstart,
                     float* __restrict__ out_loss) {
    int tid = threadIdx.x;
    int e = tid >> 5, j = tid & 31;
    __shared__ float smean[NEXP], svar[NEXP];
    float s = 0.0f, s2 = 0.0f;
    for (int t = j; t < T_TOK; t += 32) {
        float v = logits[t * NEXP + e];
        s += v; s2 += v * v;
    }
#pragma unroll
    for (int off = 16; off > 0; off >>= 1) {
        s  += __shfl_xor(s,  off, 32);
        s2 += __shfl_xor(s2, off, 32);
    }
    if (j == 0) {
        float mean = s * (1.0f / (float)T_TOK);
        float var  = (s2 - (float)T_TOK * mean * mean) / (float)(T_TOK - 1); // ddof=1
        smean[e] = mean; svar[e] = var;
    }
    __syncthreads();
    if (tid == 0) {
        float loss = 0.0f;
        for (int k = 0; k < NEXP; ++k)
            loss += svar[k] / (smean[k] * smean[k] + 1e-8f);
        *out_loss = 0.01f * loss * (1.0f / (float)NEXP);
        int off = 0, g = 0;
        for (int k = 0; k < NEXP; ++k) {
            offs[k] = off;  gstart[k] = g;
            off += count[k]; g += (count[k] + 15) >> 4;
        }
        gstart[NEXP] = g;
    }
}

// ---------------------------------------------------------------------------
// Kernel 3: build per-expert token permutation (order within expert is
// irrelevant: each token writes only its own output rows).
// ---------------------------------------------------------------------------
__global__ void moe_scatter_kernel(const int* __restrict__ sel, const int* __restrict__ offs,
                                   int* __restrict__ cursor, int* __restrict__ perm) {
    int t = blockIdx.x * blockDim.x + threadIdx.x;
    if (t < T_TOK) {
        int e = sel[t];
        int pos = atomicAdd(&cursor[e], 1);
        perm[offs[e] + pos] = t;
    }
}

// ---------------------------------------------------------------------------
// Kernel 4: fused expert FFN. One block = 16 tokens of one expert, 8 waves.
//  hmid = gelu(x @ w1[e]^T + b1[e]);  y = hmid @ w2[e]^T + b2[e]
// f32->f16 converted on the fly; WMMA f32 accumulation; register-blocked
// 4 N-tiles (phase B) / 2 N-tiles (phase C) per A-fragment load.
// ---------------------------------------------------------------------------
__global__ __launch_bounds__(256)
void moe_ffn_kernel(const float* __restrict__ x,
                    const float* __restrict__ w1, const float* __restrict__ b1,
                    const float* __restrict__ w2, const float* __restrict__ b2,
                    const int* __restrict__ perm, const int* __restrict__ count,
                    const int* __restrict__ offs, const int* __restrict__ gstart,
                    float* __restrict__ out) {
    __shared__ __align__(16) _Float16 xs[16 * HDIM];   // 24 KB
    __shared__ __align__(16) _Float16 hs[16 * IDIM];   // 96 KB
    __shared__ int tok[16];

    int g = blockIdx.x;
    if (g >= gstart[NEXP]) return;                 // uniform per block
    int e = 0;
    while (e < NEXP - 1 && g >= gstart[e + 1]) ++e;
    int chunk = g - gstart[e];

    int tid = threadIdx.x;
    if (tid < 16) {
        int local = chunk * 16 + tid;
        tok[tid] = (local < count[e]) ? perm[offs[e] + local] : -1;
    }
    __syncthreads();

    // Stage token rows into LDS as f16 (padding rows -> 0).
    for (int idx = tid; idx < 16 * HDIM; idx += 256) {
        int m = idx / HDIM, k = idx - m * HDIM;
        int t = tok[m];
        xs[idx] = (t >= 0) ? (_Float16)x[(size_t)t * HDIM + k] : (_Float16)0.0f;
    }
    __syncthreads();

    int wv = tid >> 5, lane = tid & 31;
    int nlo = lane & 15, ghi = lane >> 4;          // N index / K-half select

    // ---- Phase B: hmid[16 x IDIM], 4 i-tiles per A fragment ----
#pragma unroll 1
    for (int j = 0; j < 6; ++j) {
        int ibase = (wv + 8 * j) * 64;             // 4 tiles: ibase + 16*u
        v8f c[4];
#pragma unroll
        for (int u = 0; u < 4; ++u) c[u] = (v8f){0.f,0.f,0.f,0.f,0.f,0.f,0.f,0.f};
        const float* bp = w1 + ((size_t)e * IDIM + ibase + nlo) * HDIM + ghi * 16;
        for (int k0 = 0; k0 < HDIM; k0 += 32) {
            v16h a = load_a_frag_lds(&xs[nlo * HDIM], k0, ghi);
#pragma unroll
            for (int u = 0; u < 4; ++u) {
                v16h b = load_b_frag_f32(bp + (size_t)u * 16 * HDIM + k0);
                c[u] = __builtin_amdgcn_wmma_f32_16x16x32_f16(false, a, false, b,
                                                              (short)0, c[u], false, false);
            }
        }
#pragma unroll
        for (int u = 0; u < 4; ++u) {
            int i0 = ibase + u * 16;
            float bias = b1[e * IDIM + i0 + nlo];
#pragma unroll
            for (int r = 0; r < 8; ++r) {          // C layout: lane=N+16*(M/8), vgpr=M%8
                int m = r + 8 * ghi;
                float v = c[u][r] + bias;
                v = 0.5f * v * (1.0f + erff(v * 0.70710678118654752f)); // exact gelu
                hs[m * IDIM + i0 + nlo] = (_Float16)v;
            }
        }
    }
    __syncthreads();

    // ---- Phase C: y[16 x HDIM] = hmid @ w2[e]^T + b2[e], 2 h-tiles per A ----
#pragma unroll 1
    for (int j = 0; j < 3; ++j) {
        int hbase = (wv + 8 * j) * 32;             // 2 tiles: hbase + 16*u
        v8f c[2];
#pragma unroll
        for (int u = 0; u < 2; ++u) c[u] = (v8f){0.f,0.f,0.f,0.f,0.f,0.f,0.f,0.f};
        const float* bp = w2 + ((size_t)e * HDIM + hbase + nlo) * IDIM + ghi * 16;
        for (int k0 = 0; k0 < IDIM; k0 += 32) {
            v16h a = load_a_frag_lds(&hs[nlo * IDIM], k0, ghi);
#pragma unroll
            for (int u = 0; u < 2; ++u) {
                v16h b = load_b_frag_f32(bp + (size_t)u * 16 * IDIM + k0);
                c[u] = __builtin_amdgcn_wmma_f32_16x16x32_f16(false, a, false, b,
                                                              (short)0, c[u], false, false);
            }
        }
#pragma unroll
        for (int u = 0; u < 2; ++u) {
            int h0 = hbase + u * 16;
            float bias = b2[e * HDIM + h0 + nlo];
#pragma unroll
            for (int r = 0; r < 8; ++r) {
                int m = r + 8 * ghi;
                int t = tok[m];
                if (t >= 0)                          // gate weight == 1.0 for K=1
                    out[(size_t)t * HDIM + h0 + nlo] = c[u][r] + bias;
            }
        }
    }
}

// ---------------------------------------------------------------------------
extern "C" void kernel_launch(void* const* d_in, const int* in_sizes, int n_in,
                              void* d_out, int out_size, void* d_ws, size_t ws_size,
                              hipStream_t stream) {
    const float* x  = (const float*)d_in[0];
    const float* gw = (const float*)d_in[1];
    const float* gb = (const float*)d_in[2];
    const float* w1 = (const float*)d_in[3];
    const float* b1 = (const float*)d_in[4];
    const float* w2 = (const float*)d_in[5];
    const float* b2 = (const float*)d_in[6];
    float* out = (float*)d_out;

    // Workspace layout (~41 KB)
    int*   sel    = (int*)d_ws;          // 1024
    int*   count  = sel    + T_TOK;      // 8
    int*   offs   = count  + NEXP;       // 8
    int*   cursor = offs   + NEXP;       // 8
    int*   gstart = cursor + NEXP;       // 9 (+pad)
    int*   perm   = gstart + 16;         // 1024
    float* logits = (float*)(perm + T_TOK); // 1024*8

    moe_init_kernel<<<1, 32, 0, stream>>>(count, cursor);
    moe_gate_kernel<<<T_TOK, 32, 0, stream>>>(x, gw, gb, sel, count, logits);
    moe_plan_kernel<<<1, 256, 0, stream>>>(logits, count, offs, gstart,
                                           out + (size_t)T_TOK * HDIM);
    moe_scatter_kernel<<<4, 256, 0, stream>>>(sel, offs, cursor, perm);
    // max groups = T/16 + (E-1) = 71; launch 72, extras exit on gstart[8]
    moe_ffn_kernel<<<72, 256, 0, stream>>>(x, w1, b1, w2, b2,
                                           perm, count, offs, gstart, out);
}
